// LSTM_80367428043317
// MI455X (gfx1250) — matrix operand order repaired and checked by reading, not compiled, not run on hardware
//
#include <hip/hip_runtime.h>
#include <hip/hip_bf16.h>

// ---------------------------------------------------------------------------
// LSTM forward for B=64, S=512, I=1024, H=1024 on gfx1250 (MI455X).
//  - phase 1: f32->f16 casts of inputs & weights (weights are already in the
//             N-major [4096 x 1024] layout WMMA's B-fragment wants)
//  - phase 2: x_proj = X @ Wi^T + (b_i + b_h) via v_wmma_f32_16x16x32_f16
//  - phase 3: 512 sequential fused step kernels: z = x_proj[s] + h @ Wh^T
//             (WMMA), gates + cell update fused via LDS, h double-buffered.
//  v6: pin the software pipeline against the pressure-first scheduler:
//      (a) amdgpu_waves_per_eu(2,8) relaxes the VGPR budget so two fragment
//          generations + accumulators fit without spilling/sinking;
//      (b) __builtin_amdgcn_sched_barrier(0) fences each generation's load
//          block from the previous generation's WMMA block, so loads stay
//          in flight across the WMMAs (partial s_wait_loadcnt, not 0x0).
// ---------------------------------------------------------------------------

typedef __attribute__((ext_vector_type(16))) _Float16 v16h;
typedef __attribute__((ext_vector_type(8)))  float    v8f;
typedef __attribute__((ext_vector_type(4)))  float    f32x4;

union Frag16 { v16h h; f32x4 q[2]; };

#define LSTM_B   64
#define LSTM_S   512
#define LSTM_I   1024
#define LSTM_H   1024
#define LSTM_N4  4096          // 4 gates * H
#define LSTM_M   (LSTM_B * LSTM_S)

// ---- WMMA fragment loaders (wave32 layouts per CDNA5 ISA 7.12.2) ----------

// A: 16x32 f16 tile from row-major [rows x ld] storage, tile origin p0.
// lane<16 : row=lane,    K = {k..k+7, k+16..k+23}
// lane>=16: row=lane-16, K = {k+8..k+15, k+24..k+31}
__device__ __forceinline__ v16h load_a16x32(const _Float16* p0, int ld, int lane) {
  const int m  = lane & 15;
  const int kk = (lane & 16) ? 8 : 0;
  const _Float16* p = p0 + (size_t)m * ld + kk;
  Frag16 f;
  f.q[0] = *(const f32x4*)(p);        // 8 halves
  f.q[1] = *(const f32x4*)(p + 16);   // 8 halves, +16 elements in K
  return f.h;
}

// B: 32x16 f16 tile, stored N-major [N x ld] (i.e. already "transposed"),
// tile origin p0 = &W[n0*ld + k0].
// lane<16 : col=lane,   K = {k..k+7, k+16..k+23} ; lane>=16: col=lane-16, K = {k+8..k+15, k+24..k+31}
__device__ __forceinline__ v16h load_bT32x16(const _Float16* p0, int ld, int lane) {
  const int n  = lane & 15;
  const int kk = (lane & 16) ? 8 : 0;
  const _Float16* p = p0 + (size_t)n * ld + kk;
  Frag16 f;
  f.q[0] = *(const f32x4*)(p);
  f.q[1] = *(const f32x4*)(p + 16);
  return f.h;
}

__device__ __forceinline__ v8f wmma_f16(v16h a, v16h b, v8f c) {
  return __builtin_amdgcn_wmma_f32_16x16x32_f16(false, a, false, b,
                                                (short)0, c, false, false);
}

__device__ __forceinline__ float sigmoid_f(float x) {
  return 1.0f / (1.0f + __expf(-x));
}

// ---- elementwise cast ------------------------------------------------------

__global__ void cast_f32_to_f16_kernel(const float* __restrict__ src,
                                       _Float16* __restrict__ dst, size_t n) {
  size_t i = (size_t)blockIdx.x * blockDim.x + threadIdx.x;
  if (i < n) dst[i] = (_Float16)src[i];
}

// bias fuse + state init
__global__ void prep_kernel(const float* __restrict__ b_i,
                            const float* __restrict__ b_h,
                            float* __restrict__ bias,
                            const float* __restrict__ h0,
                            const float* __restrict__ c0,
                            _Float16* __restrict__ h16,
                            float* __restrict__ Cst) {
  int i = blockIdx.x * blockDim.x + threadIdx.x;
  if (i < LSTM_N4) bias[i] = b_i[i] + b_h[i];
  if (i < LSTM_B * LSTM_H) {
    h16[i] = (_Float16)h0[i];
    Cst[i] = c0[i];
  }
}

// ---- x_proj GEMM: [M x 1024] @ [1024 x 4096] (+bias) -> [M x 4096] --------
// 256-thread WG = 8 waves. WG tile: 128 (M) x 64 (N). Each wave: 16 x 64,
// i.e. 4 accumulators, A reused 4x per generation. Fully unrolled, 2-slot
// pipeline pinned with sched_barriers.
__global__ void __launch_bounds__(256)
__attribute__((amdgpu_waves_per_eu(2, 8)))
xproj_gemm_kernel(const _Float16* __restrict__ X,     // [M x 1024]
                  const _Float16* __restrict__ WiT,   // [4096 x 1024] N-major
                  const float*    __restrict__ bias,  // [4096]
                  float*          __restrict__ P) {   // [M x 4096]
  const int K = LSTM_I;
  const int NGEN = LSTM_I / 32;                       // 32 fragment generations
  const int lane = threadIdx.x & 31;
  const int wid  = threadIdx.x >> 5;                  // 0..7
  const int m0 = blockIdx.x * 128 + wid * 16;
  const int n0 = blockIdx.y * 64;

  v8f acc[4];
  #pragma unroll
  for (int t = 0; t < 4; ++t) acc[t] = (v8f)(0.0f);

  const _Float16* Arow = X + (size_t)m0 * K;
  const _Float16* Bcol = WiT + (size_t)n0 * K;

  // slot-based pipeline, depth 1: all indices compile-time constants
  v16h aS[2];
  v16h bS[2][4];
  aS[0] = load_a16x32(Arow, K, lane);
  #pragma unroll
  for (int t = 0; t < 4; ++t)
    bS[0][t] = load_bT32x16(Bcol + (size_t)t * 16 * K, K, lane);

  #pragma unroll
  for (int g = 0; g < NGEN; ++g) {
    const int cur = g & 1;
    const int nxt = cur ^ 1;
    if (g + 1 < NGEN) {
      const int kn = (g + 1) * 32;
      aS[nxt] = load_a16x32(Arow + kn, K, lane);
      #pragma unroll
      for (int t = 0; t < 4; ++t)
        bS[nxt][t] = load_bT32x16(Bcol + (size_t)t * 16 * K + kn, K, lane);
    }
    // fence: next-gen loads may not sink below this generation's WMMAs
    __builtin_amdgcn_sched_barrier(0);
    #pragma unroll
    for (int t = 0; t < 4; ++t)
      acc[t] = wmma_f16(aS[cur], bS[cur][t], acc[t]);
  }

  // C layout: VGPR r, lane<16 -> (m=r, n=lane); lane>=16 -> (m=8+r, n=lane-16)
  const int nc0 = lane & 15;
  const int mo  = (lane & 16) ? 8 : 0;
  #pragma unroll
  for (int t = 0; t < 4; ++t) {
    const int nc = n0 + t * 16 + nc0;
    const float bv = bias[nc];
    #pragma unroll
    for (int r = 0; r < 8; ++r)
      P[(size_t)(m0 + mo + r) * LSTM_N4 + nc] = acc[t][r] + bv;
  }
}

// ---- one recurrence step ---------------------------------------------------
// grid (B/16, H/16) = (4, 64); 128 threads = 4 waves, wave g computes gate g's
// 16x16 z-tile: z = Hprev @ Wh^T + x_proj. Gates fused through LDS.
// Fully unrolled, 3-slot (depth-2) pipeline pinned with sched_barriers; this
// is the latency-critical sequential part: keep >=4 b128 loads in flight.
__global__ void __launch_bounds__(128)
__attribute__((amdgpu_waves_per_eu(2, 8)))
lstm_step_kernel(const _Float16* __restrict__ Hprev16,  // [64 x 1024]
                 const _Float16* __restrict__ WhT,      // [4096 x 1024] N-major
                 const float*    __restrict__ P,        // [M x 4096]
                 float*          __restrict__ Cst,      // [64 x 1024]
                 _Float16*       __restrict__ Hnext16,  // [64 x 1024]
                 float*          __restrict__ Out,      // full d_out
                 int s) {
  const int K = LSTM_H;
  const int NGEN = LSTM_H / 32;
  __shared__ float zs[4][16][16];

  const int lane = threadIdx.x & 31;
  const int g    = threadIdx.x >> 5;                   // gate id 0..3 (i,f,g,o)
  const int b0   = blockIdx.x * 16;
  const int o0   = blockIdx.y * 16;

  v8f acc = (v8f)(0.0f);
  const _Float16* Abase = Hprev16 + (size_t)b0 * K;
  const _Float16* Bbase = WhT + (size_t)(g * LSTM_H + o0) * K;

  v16h aS[3], bS[3];
  aS[0] = load_a16x32(Abase, K, lane);
  bS[0] = load_bT32x16(Bbase, K, lane);
  aS[1] = load_a16x32(Abase + 32, K, lane);
  bS[1] = load_bT32x16(Bbase + 32, K, lane);

  #pragma unroll
  for (int q = 0; q < NGEN; ++q) {
    if (q + 2 < NGEN) {
      const int kn = (q + 2) * 32;
      if ((kn & 255) == 0)
        __builtin_prefetch(Bbase + kn + 256, 0, 3);    // global_prefetch_b8
      aS[(q + 2) % 3] = load_a16x32(Abase + kn, K, lane);
      bS[(q + 2) % 3] = load_bT32x16(Bbase + kn, K, lane);
    }
    // fence: lookahead loads may not sink below this generation's WMMA
    __builtin_amdgcn_sched_barrier(0);
    acc = wmma_f16(aS[q % 3], bS[q % 3], acc);
  }

  // z = acc + x_proj ; stage per-gate tile in LDS
  const int nc = lane & 15;
  const int mo = (lane & 16) ? 8 : 0;
  #pragma unroll
  for (int r = 0; r < 8; ++r) {
    const int b = b0 + mo + r;
    const float z = acc[r] +
        P[((size_t)b * LSTM_S + s) * LSTM_N4 + g * LSTM_H + o0 + nc];
    zs[g][mo + r][nc] = z;
  }
  __syncthreads();

  // fused gates + cell update; each WG exclusively owns its (b,o) block
  for (int idx = threadIdx.x; idx < 256; idx += 128) {
    const int m = idx >> 4, n = idx & 15;
    const int b = b0 + m, o = o0 + n;
    const float it = sigmoid_f(zs[0][m][n]);
    const float ft = sigmoid_f(zs[1][m][n]);
    const float gt = tanhf(zs[2][m][n]);
    const float ot = sigmoid_f(zs[3][m][n]);
    const size_t co = (size_t)b * LSTM_H + o;
    const float cn = ft * Cst[co] + it * gt;
    Cst[co] = cn;
    const float hn = ot * tanhf(cn);
    Hnext16[co] = (_Float16)hn;
    Out[((size_t)b * LSTM_S + s) * LSTM_H + o] = hn;         // x output
    const size_t tail = (size_t)LSTM_B * LSTM_S * LSTM_H;
    Out[tail + co] = hn;                                     // h_last (last write wins)
    Out[tail + (size_t)LSTM_B * LSTM_H + co] = cn;           // c_last
  }
}

// ---------------------------------------------------------------------------

extern "C" void kernel_launch(void* const* d_in, const int* in_sizes, int n_in,
                              void* d_out, int out_size, void* d_ws, size_t ws_size,
                              hipStream_t stream) {
  const float* x    = (const float*)d_in[0];   // (B,S,I)
  const float* h0   = (const float*)d_in[1];   // (1,B,H)
  const float* c0   = (const float*)d_in[2];   // (1,B,H)
  const float* w_i  = (const float*)d_in[3];   // (4,H,I) == N-major [4096 x 1024]
  const float* w_h  = (const float*)d_in[4];   // (4,H,H) == N-major [4096 x 1024]
  const float* b_i  = (const float*)d_in[5];   // (4,H)
  const float* b_h  = (const float*)d_in[6];   // (4,H)
  float* out = (float*)d_out;

  char* ws = (char*)d_ws;
  size_t off = 0;
  _Float16* x16  = (_Float16*)(ws + off); off += (size_t)LSTM_M * LSTM_I * 2;      // 64 MB
  _Float16* wi16 = (_Float16*)(ws + off); off += (size_t)LSTM_N4 * LSTM_I * 2;     // 8 MB
  _Float16* wh16 = (_Float16*)(ws + off); off += (size_t)LSTM_N4 * LSTM_H * 2;     // 8 MB
  float*    bias = (float*)   (ws + off); off += (size_t)LSTM_N4 * 4;
  float*    Cst  = (float*)   (ws + off); off += (size_t)LSTM_B * LSTM_H * 4;
  _Float16* hA   = (_Float16*)(ws + off); off += (size_t)LSTM_B * LSTM_H * 2;
  _Float16* hB   = (_Float16*)(ws + off); off += (size_t)LSTM_B * LSTM_H * 2;
  float*    P    = (float*)   (ws + off); off += (size_t)LSTM_M * LSTM_N4 * 4;     // 512 MB

  // phase 1: casts + bias fuse + state init
  {
    size_t n = (size_t)LSTM_M * LSTM_I;
    cast_f32_to_f16_kernel<<<dim3((unsigned)((n + 255) / 256)), 256, 0, stream>>>(x, x16, n);
    n = (size_t)LSTM_N4 * LSTM_I;
    cast_f32_to_f16_kernel<<<dim3((unsigned)((n + 255) / 256)), 256, 0, stream>>>(w_i, wi16, n);
    n = (size_t)LSTM_N4 * LSTM_H;
    cast_f32_to_f16_kernel<<<dim3((unsigned)((n + 255) / 256)), 256, 0, stream>>>(w_h, wh16, n);
    prep_kernel<<<dim3((LSTM_B * LSTM_H + 255) / 256), 256, 0, stream>>>(
        b_i, b_h, bias, h0, c0, hA, Cst);
  }

  // phase 2: x_proj GEMM (M=32768, N=4096, K=1024)
  xproj_gemm_kernel<<<dim3(LSTM_M / 128, LSTM_N4 / 64), 256, 0, stream>>>(
      x16, wi16, bias, P);

  // phase 3: sequential scan, ping-pong h buffers
  for (int s = 0; s < LSTM_S; ++s) {
    const _Float16* hp = (s & 1) ? hB : hA;
    _Float16*       hn = (s & 1) ? hA : hB;
    lstm_step_kernel<<<dim3(LSTM_B / 16, LSTM_H / 16), 128, 0, stream>>>(
        hp, wh16, P, Cst, hn, out, s);
  }
}
